// MultiHeadAttention_48249662603327
// MI455X (gfx1250) — matrix operand order repaired and checked by reading, not compiled, not run on hardware
//
#include <hip/hip_runtime.h>
#include <hip/hip_bf16.h>

// ---------------------------------------------------------------------------
// MHA block (B=1, S=4096, E=768, H=12, Dh=64) for gfx1250 (MI455X).
// All matmuls on v_wmma_f32_16x16x32_bf16 (fp32 accumulate).
// Tile staging via Tensor Data Mover (tensor_load_to_lds), double-buffered,
// with TDM hardware padding providing the LDS leading dimension.
// Pipeline: cvt/transpose -> QKV GEMM -> RoPE scatter -> flash attn -> out GEMM.
// ---------------------------------------------------------------------------

#define SEQ   4096
#define EMB   768
#define HEADS 12
#define DH    64
#define QKV3  2304   // 3*EMB

typedef __bf16 v16bf __attribute__((ext_vector_type(16)));
typedef float  v8f   __attribute__((ext_vector_type(8)));
typedef unsigned int v4u __attribute__((ext_vector_type(4)));
typedef int v4i __attribute__((ext_vector_type(4)));
typedef int v8i __attribute__((ext_vector_type(8)));

union Frag {
    v16bf          bf;
    unsigned short u16[16];
    unsigned int   u32[8];
};

__device__ __forceinline__ unsigned short f2bf(float f) {
    union { float f; unsigned u; } v; v.f = f;
    unsigned r = v.u + 0x7FFFu + ((v.u >> 16) & 1u);   // round-to-nearest-even
    return (unsigned short)(r >> 16);
}

__device__ __forceinline__ v8f wmma_bf16(const Frag& a, const Frag& b, v8f c) {
    return __builtin_amdgcn_wmma_f32_16x16x32_bf16(
        /*neg_a=*/false, a.bf, /*neg_b=*/false, b.bf,
        /*c_mod=*/(short)0, c, /*reuse_a=*/false, /*reuse_b=*/false);
}

// A-frag 16x32 bf16 from row-major [row][k] storage (k pairs contiguous).
// Lane row = lane&15; 16 consecutive K, base = (lane>=16)*16 (matches B-frag).
__device__ __forceinline__ Frag load_a16x32(const unsigned short* p,
                                            int row0, int k0, int ld) {
    Frag f;
    const int lane = threadIdx.x & 31;
    const unsigned short* r = p + (size_t)(row0 + (lane & 15)) * ld
                                + k0 + (lane >> 4) * 16;
#pragma unroll
    for (int v = 0; v < 8; ++v) {
        f.u32[v] = *(const unsigned int*)(r + 2 * v);
    }
    return f;
}

// B-frag 32x16 bf16 from n-major storage [n][k] (k pairs contiguous -> dwords).
// Lane holds column n = lane&15; 16 consecutive K, base = (lane>=16)*16 (same order as A-frag).
__device__ __forceinline__ Frag load_b32x16_nmajor(const unsigned short* p,
                                                   int k0, int n0, int ld) {
    Frag f;
    const int lane = threadIdx.x & 31;
    const unsigned short* r = p + (size_t)(n0 + (lane & 15)) * ld
                                + k0 + (lane >> 4) * 16;
#pragma unroll
    for (int v = 0; v < 8; ++v)
        f.u32[v] = *(const unsigned int*)(r + 2 * v);
    return f;
}

// ---------------------------------------------------------------------------
// Tensor Data Mover: 2D tile (tile_w x tile_h elements, 2B each) from a
// row-major tensor (row_stride elements) into LDS at lds_off, with optional
// per-row LDS padding. D# layout per CDNA5 ISA ch.8 (group2/3 zero => 2D).
// Toolchain uses the 6-arg builtin (extra int32x8 group before cpol).
// ---------------------------------------------------------------------------
__device__ __forceinline__ void tdm_load_2d(unsigned lds_off, const void* gptr,
                                            unsigned tile_w, unsigned tile_h,
                                            unsigned tensor_w, unsigned tensor_h,
                                            unsigned row_stride,
                                            unsigned pad_int, unsigned pad_amt,
                                            unsigned pad_en) {
    const unsigned long long ga = (unsigned long long)gptr;
    const v4u g0 = {
        1u,                                                   // count=1, user D#
        lds_off,                                              // lds_addr
        (unsigned)(ga & 0xffffffffull),                       // global_addr[31:0]
        (unsigned)((ga >> 32) & 0x1ffffffull) | (2u << 30)    // addr[56:32]|type=2
    };
    const v8i g1 = {
        (int)((1u << 16) | (pad_en << 20) | (pad_int << 22) | (pad_amt << 25)),
        (int)((tensor_w & 0xffffu) << 16),                     // dim0[15:0]@48
        (int)(((tensor_w >> 16) & 0xffffu) | ((tensor_h & 0xffffu) << 16)),
        (int)(((tensor_h >> 16) & 0xffffu) | ((tile_w & 0xffffu) << 16)),
        (int)(tile_h & 0xffffu),                               // tile_dim1
        (int)row_stride,                                       // dim0_stride lo
        0, 0
    };
    const v4i zero4 = {0, 0, 0, 0};
    const v8i zero8 = {0, 0, 0, 0, 0, 0, 0, 0};
    __builtin_amdgcn_tensor_load_to_lds(g0, g1, zero4, zero4, zero8, 0);
}

// ---------------------------------------------------------------------------
// Kernel: fp32 -> bf16
// ---------------------------------------------------------------------------
__global__ __launch_bounds__(256)
void cvt_f32_bf16(const float* __restrict__ in, unsigned short* __restrict__ out, int n) {
    const int i = blockIdx.x * 256 + threadIdx.x;
    if (i < n) out[i] = f2bf(in[i]);
}

// Kernel: out[n][k] = bf16(in[k][n])  (weights -> transposed bf16)
__global__ __launch_bounds__(256)
void transpose_f32_bf16(const float* __restrict__ in, unsigned short* __restrict__ out,
                        int rows, int cols) {
    const int idx = blockIdx.x * 256 + threadIdx.x;
    if (idx >= rows * cols) return;
    const int k = idx / cols, n = idx % cols;
    out[(size_t)n * rows + k] = f2bf(in[idx]);
}

// ---------------------------------------------------------------------------
// Kernel: C(f32, MxN) = A(bf16, [M][K]) @ Bt(bf16, [N][K])^T + bias[N]
// Block tile 128x64, 256 threads = 8 waves (4x2 of 32x32), K-step 32.
// TDM double-buffered staging; pad: 1 dword per 16 dwords -> LDS ld = 34.
// ---------------------------------------------------------------------------
__global__ __launch_bounds__(256)
void gemm_bf16_tn(const unsigned short* __restrict__ A,
                  const unsigned short* __restrict__ Bt,
                  const float* __restrict__ bias,
                  float* __restrict__ C,
                  int M, int N, int K) {
    __shared__ unsigned short As[2][128 * 34];
    __shared__ unsigned short Bs[2][64 * 34];

    const int tid  = threadIdx.x;
    const int m0   = blockIdx.y * 128;
    const int n0   = blockIdx.x * 64;
    const int w    = tid >> 5;
    const int wr   = (w & 3) * 32;
    const int wc   = (w >> 2) * 32;
    const int lane = tid & 31;

    auto issue = [&](int k0, int buf) {
        tdm_load_2d((unsigned)(size_t)&As[buf][0], A + (size_t)m0 * K + k0,
                    32, 128, (unsigned)K, (unsigned)M, (unsigned)K, 3, 0, 1);
        tdm_load_2d((unsigned)(size_t)&Bs[buf][0], Bt + (size_t)n0 * K + k0,
                    32, 64, (unsigned)K, (unsigned)N, (unsigned)K, 3, 0, 1);
    };

    const v8f zero = {0.f, 0.f, 0.f, 0.f, 0.f, 0.f, 0.f, 0.f};
    v8f acc[2][2];
    acc[0][0] = zero; acc[0][1] = zero; acc[1][0] = zero; acc[1][1] = zero;

    if (w == 0) issue(0, 0);

    const int nk = K >> 5;
    for (int t = 0; t < nk; ++t) {
        const int buf = t & 1;
        __syncthreads();                       // prior reads of buf^1 done
        if (w == 0) {
            if (t + 1 < nk) {
                issue((t + 1) << 5, buf ^ 1);
                __builtin_amdgcn_s_wait_tensorcnt(2);   // tile t complete
            } else {
                __builtin_amdgcn_s_wait_tensorcnt(0);
            }
        }
        __syncthreads();

        const Frag a0 = load_a16x32(&As[buf][0], wr,      0, 34);
        const Frag a1 = load_a16x32(&As[buf][0], wr + 16, 0, 34);
        const Frag b0 = load_b32x16_nmajor(&Bs[buf][0], 0, wc,      34);
        const Frag b1 = load_b32x16_nmajor(&Bs[buf][0], 0, wc + 16, 34);
        acc[0][0] = wmma_bf16(a0, b0, acc[0][0]);
        acc[0][1] = wmma_bf16(a0, b1, acc[0][1]);
        acc[1][0] = wmma_bf16(a1, b0, acc[1][0]);
        acc[1][1] = wmma_bf16(a1, b1, acc[1][1]);
    }

    // Epilogue: C-frag element v at (row = v + 8*(lane>=16), col = lane&15).
    const int cl = lane & 15, half = lane >> 4;
    const float b0v = bias[n0 + wc + cl];
    const float b1v = bias[n0 + wc + 16 + cl];
    const size_t base = (size_t)(m0 + wr + 8 * half) * N + (n0 + wc + cl);
#pragma unroll
    for (int mi = 0; mi < 2; ++mi)
#pragma unroll
        for (int v = 0; v < 8; ++v) {
            const size_t ro = base + (size_t)(mi * 16 + v) * N;
            C[ro]      = acc[mi][0][v] + b0v;
            C[ro + 16] = acc[mi][1][v] + b1v;
        }
}

// ---------------------------------------------------------------------------
// Kernel: RoPE + scatter qkv(f32 [S][2304]) ->
//   Q,K roped bf16 [H][S][64];  V bf16 transposed [H][64][S].
// ---------------------------------------------------------------------------
__global__ __launch_bounds__(256)
void rope_scatter(const float* __restrict__ qkv,
                  unsigned short* __restrict__ Qo,
                  unsigned short* __restrict__ Ko,
                  unsigned short* __restrict__ Vt) {
    const int idx = blockIdx.x * 256 + threadIdx.x;
    if (idx >= SEQ * EMB) return;
    const int s = idx / EMB;
    const int c = idx % EMB;
    const int h = c >> 6;
    const int d = c & 63;

    const float* row = qkv + (size_t)s * QKV3;
    const float q  = row[c];
    const float k  = row[EMB + c];
    const float vv = row[2 * EMB + c];

    const int   i   = d & 31;                               // pair index
    const float ang = (float)s * __expf(-0.28782313663f * (float)i); // ln(1e4)/32
    float cs, sn;
    __sincosf(ang, &cs, &sn);

    float qr, kr;
    if (d < 32) { qr = -row[c + 32];  kr = -row[EMB + c + 32]; }
    else        { qr =  row[c - 32];  kr =  row[EMB + c - 32]; }

    const size_t o = ((size_t)h * SEQ + s) * DH + d;
    Qo[o] = f2bf(q * cs + qr * sn);
    Ko[o] = f2bf(k * cs + kr * sn);
    Vt[((size_t)h * DH + d) * SEQ + s] = f2bf(vv);
}

// ---------------------------------------------------------------------------
// Kernel: causal flash attention per head.
// Grid (H, S/64); 128 threads = 4 waves, each wave owns 16 query rows.
// K tile [key][d] and V^T tile [d][key] TDM'd into LDS (ld=66), double-buffered.
// Softmax stats live in registers aligned with the WMMA C-frag row layout.
// ---------------------------------------------------------------------------
__global__ __launch_bounds__(128)
void attn_flash(const unsigned short* __restrict__ Q,
                const unsigned short* __restrict__ Kt,
                const unsigned short* __restrict__ Vt,
                unsigned short* __restrict__ O) {
    __shared__ unsigned short Ks[2][64 * 66];    // [key][d], padded
    __shared__ unsigned short Vs[2][64 * 66];    // [d][key], padded
    __shared__ unsigned short Ps[4][16 * 64];    // per-wave P strip [row][key]

    const int h    = blockIdx.x;
    const int q0   = blockIdx.y * 64;
    const int tid  = threadIdx.x;
    const int w    = tid >> 5;
    const int lane = tid & 31;
    const int half = lane >> 4;
    const int cl   = lane & 15;

    const unsigned short* Qh  = Q  + (size_t)h * SEQ * DH;
    const unsigned short* Kh  = Kt + (size_t)h * SEQ * DH;
    const unsigned short* Vth = Vt + (size_t)h * DH * SEQ;

    auto issue = [&](int j, int buf) {
        tdm_load_2d((unsigned)(size_t)&Ks[buf][0], Kh + (size_t)j * 64 * DH,
                    64, 64, DH, SEQ, DH, 4, 0, 1);            // pad 1dw / 32dw
        tdm_load_2d((unsigned)(size_t)&Vs[buf][0], Vth + (size_t)j * 64,
                    64, 64, SEQ, DH, SEQ, 4, 0, 1);
    };

    // Q frags stay in registers the whole kernel.
    const Frag qa0 = load_a16x32(Qh, q0 + w * 16, 0,  DH);
    const Frag qa1 = load_a16x32(Qh, q0 + w * 16, 32, DH);

    const v8f zero = {0.f, 0.f, 0.f, 0.f, 0.f, 0.f, 0.f, 0.f};
    v8f   o[4];
    float m[8], l[8];
#pragma unroll
    for (int c = 0; c < 4; ++c) o[c] = zero;
#pragma unroll
    for (int v = 0; v < 8; ++v) { m[v] = -3.0e38f; l[v] = 0.f; }

    const int jmax = q0 >> 6;
    if (w == 0) issue(0, 0);

    for (int j = 0; j <= jmax; ++j) {
        const int buf = j & 1;
        __syncthreads();                       // prior reads of buf^1 done
        if (w == 0) {
            if (j < jmax) {
                issue(j + 1, buf ^ 1);
                __builtin_amdgcn_s_wait_tensorcnt(2);   // tile j complete
            } else {
                __builtin_amdgcn_s_wait_tensorcnt(0);
            }
        }
        __syncthreads();

        // Scores S = (Q @ K^T) * 1/sqrt(Dh); Ks is [n=key][k=d].
        v8f s[4];
#pragma unroll
        for (int c = 0; c < 4; ++c) {
            s[c] = zero;
            s[c] = wmma_bf16(qa0, load_b32x16_nmajor(&Ks[buf][0], 0,  c * 16, 66), s[c]);
            s[c] = wmma_bf16(qa1, load_b32x16_nmajor(&Ks[buf][0], 32, c * 16, 66), s[c]);
        }

        // Causal mask + scale + per-row tile max.
        float mt[8];
#pragma unroll
        for (int v = 0; v < 8; ++v) mt[v] = -3.0e38f;
#pragma unroll
        for (int c = 0; c < 4; ++c)
#pragma unroll
            for (int v = 0; v < 8; ++v) {
                const int col = j * 64 + c * 16 + cl;
                const int row = q0 + w * 16 + v + 8 * half;
                float val = s[c][v] * 0.125f;               // 1/sqrt(64)
                val = (col <= row) ? val : -3.0e38f;
                s[c][v] = val;
                mt[v] = fmaxf(mt[v], val);
            }
#pragma unroll
        for (int v = 0; v < 8; ++v) {
            mt[v] = fmaxf(mt[v], __shfl_xor(mt[v], 1));
            mt[v] = fmaxf(mt[v], __shfl_xor(mt[v], 2));
            mt[v] = fmaxf(mt[v], __shfl_xor(mt[v], 4));
            mt[v] = fmaxf(mt[v], __shfl_xor(mt[v], 8));
        }

        float alpha[8], rs[8];
#pragma unroll
        for (int v = 0; v < 8; ++v) {
            const float mn = fmaxf(m[v], mt[v]);
            alpha[v] = __expf(m[v] - mn);
            m[v] = mn;
            rs[v] = 0.f;
        }

        // P = exp(S - m); stash bf16 P in per-wave LDS strip (C->A reshape).
#pragma unroll
        for (int c = 0; c < 4; ++c)
#pragma unroll
            for (int v = 0; v < 8; ++v) {
                const float p = __expf(s[c][v] - m[v]);
                rs[v] += p;
                Ps[w][(v + 8 * half) * 64 + c * 16 + cl] = f2bf(p);
            }
#pragma unroll
        for (int v = 0; v < 8; ++v) {
            rs[v] += __shfl_xor(rs[v], 1);
            rs[v] += __shfl_xor(rs[v], 2);
            rs[v] += __shfl_xor(rs[v], 4);
            rs[v] += __shfl_xor(rs[v], 8);
            l[v] = l[v] * alpha[v] + rs[v];
        }
#pragma unroll
        for (int c = 0; c < 4; ++c)
#pragma unroll
            for (int v = 0; v < 8; ++v) o[c][v] *= alpha[v];

        // O += P @ V; Vs is [n=d][k=key]. Same-wave DS ops are in order.
        const Frag pa0 = load_a16x32(&Ps[w][0], 0, 0,  64);
        const Frag pa1 = load_a16x32(&Ps[w][0], 0, 32, 64);
#pragma unroll
        for (int c = 0; c < 4; ++c) {
            o[c] = wmma_bf16(pa0, load_b32x16_nmajor(&Vs[buf][0], 0,  c * 16, 66), o[c]);
            o[c] = wmma_bf16(pa1, load_b32x16_nmajor(&Vs[buf][0], 32, c * 16, 66), o[c]);
        }
    }

    // Normalize, write attention output re-packed to [S][E] bf16.
#pragma unroll
    for (int c = 0; c < 4; ++c)
#pragma unroll
        for (int v = 0; v < 8; ++v) {
            const int row = q0 + w * 16 + v + 8 * half;
            const int col = h * DH + c * 16 + cl;
            O[(size_t)row * EMB + col] = f2bf(o[c][v] / l[v]);
        }
}

// ---------------------------------------------------------------------------
// Host side
// ---------------------------------------------------------------------------
extern "C" void kernel_launch(void* const* d_in, const int* in_sizes, int n_in,
                              void* d_out, int out_size, void* d_ws, size_t ws_size,
                              hipStream_t stream) {
    const float* x    = (const float*)d_in[0];
    // d_in[1] = causal mask (implicit in the kernel, unused)
    const float* Wqkv = (const float*)d_in[2];
    const float* bqkv = (const float*)d_in[3];
    const float* Wout = (const float*)d_in[4];
    const float* bout = (const float*)d_in[5];
    float*       out  = (float*)d_out;

    char* ws = (char*)d_ws;
    size_t off = 0;
    unsigned short* x_bf     = (unsigned short*)(ws + off); off += (size_t)SEQ * EMB * 2;
    unsigned short* wqkvT_bf = (unsigned short*)(ws + off); off += (size_t)EMB * QKV3 * 2;
    unsigned short* woutT_bf = (unsigned short*)(ws + off); off += (size_t)EMB * EMB * 2;
    float*          qkv_f    = (float*)(ws + off);          off += (size_t)SEQ * QKV3 * 4;
    unsigned short* Qb       = (unsigned short*)(ws + off); off += (size_t)HEADS * SEQ * DH * 2;
    unsigned short* Kb       = (unsigned short*)(ws + off); off += (size_t)HEADS * SEQ * DH * 2;
    unsigned short* Vtb      = (unsigned short*)(ws + off); off += (size_t)HEADS * SEQ * DH * 2;
    unsigned short* Ob       = (unsigned short*)(ws + off); off += (size_t)SEQ * EMB * 2;
    (void)off; (void)ws_size; (void)in_sizes; (void)n_in; (void)out_size;

    // 1) One-time conversions: x -> bf16; weights -> transposed bf16 [N][K].
    cvt_f32_bf16<<<(SEQ * EMB + 255) / 256, 256, 0, stream>>>(x, x_bf, SEQ * EMB);
    transpose_f32_bf16<<<(EMB * QKV3 + 255) / 256, 256, 0, stream>>>(Wqkv, wqkvT_bf, EMB, QKV3);
    transpose_f32_bf16<<<(EMB * EMB + 255) / 256, 256, 0, stream>>>(Wout, woutT_bf, EMB, EMB);

    // 2) QKV projection: [4096x768] @ [768x2304] + bqkv -> fp32.
    gemm_bf16_tn<<<dim3(QKV3 / 64, SEQ / 128), 256, 0, stream>>>(
        x_bf, wqkvT_bf, bqkv, qkv_f, SEQ, QKV3, EMB);

    // 3) RoPE + scatter to per-head bf16 Q/K (roped) and V^T.
    rope_scatter<<<(SEQ * EMB + 255) / 256, 256, 0, stream>>>(qkv_f, Qb, Kb, Vtb);

    // 4) Causal flash attention.
    attn_flash<<<dim3(HEADS, SEQ / 64), 128, 0, stream>>>(Qb, Kb, Vtb, Ob);

    // 5) Output projection: [4096x768] @ [768x768] + bout -> d_out fp32.
    gemm_bf16_tn<<<dim3(EMB / 64, SEQ / 128), 256, 0, stream>>>(
        Ob, woutT_bf, bout, out, SEQ, EMB, EMB);
}
